// BCAModule_2680059593271
// MI455X (gfx1250) — matrix operand order repaired and checked by reading, not compiled, not run on hardware
//
#include <hip/hip_runtime.h>

// ---------------- types ----------------
typedef __attribute__((ext_vector_type(16))) __bf16 v16bf;
typedef __attribute__((ext_vector_type(8)))  __bf16 v8bf;
typedef __attribute__((ext_vector_type(4)))  __bf16 v4bf;
typedef __attribute__((ext_vector_type(8)))  float  v8f;

#define NPIX   16384   // 128*128
#define CMID   64
#define CXIN   720
#define MKEYS  1024    // 32*32 pooled
#define TILE_N 128

// Tensor Data Mover availability (arity differs between toolchains)
#if defined(__has_builtin)
#  if __has_builtin(__builtin_amdgcn_tensor_load_to_lds)
#    define BCA_TDM 1
#  endif
#endif
#ifndef BCA_TDM
#  define BCA_TDM 0
#endif

__device__ __forceinline__ __bf16 f2bf(float f) {
  unsigned u = __builtin_bit_cast(unsigned, f);
  unsigned r = (u + 0x7FFFu + ((u >> 16) & 1u)) >> 16;
  return __builtin_bit_cast(__bf16, (unsigned short)r);
}
__device__ __forceinline__ float bf2f(__bf16 b) {
  unsigned u = ((unsigned)__builtin_bit_cast(unsigned short, b)) << 16;
  return __builtin_bit_cast(float, u);
}
// Build a 16x32 A-frag / 32x16 B-frag register set from two 16B chunks.
__device__ __forceinline__ v16bf frag16(const __bf16* p0, const __bf16* p1) {
  union { v16bf v; v8bf h[2]; } u;
  u.h[0] = *(const v8bf*)p0;
  u.h[1] = *(const v8bf*)p1;
  return u.v;
}

#if BCA_TDM
typedef __attribute__((ext_vector_type(4))) unsigned v4u;
typedef __attribute__((ext_vector_type(8))) int      v8i;
typedef __attribute__((ext_vector_type(4))) int      v4i;

// Issue a TDM 2D (or 1D if tile_d1==0) tile load, global -> LDS.
// Descriptor bit layout per CDNA5 ISA 8.3/8.4 (D# groups 0/1; groups 2/3 zero).
__device__ __forceinline__ void bca_tdm_load(unsigned lds_off, const void* gptr,
                                             unsigned tile_d0, unsigned tile_d1,
                                             unsigned tens_d0, unsigned tens_d1,
                                             unsigned long long stride0_elems)
{
  const unsigned long long ga = (unsigned long long)gptr;
  v4u g0;
  g0[0] = 1u;                                                 // count=1, user D#
  g0[1] = lds_off;                                            // lds_addr (bytes)
  g0[2] = (unsigned)ga;                                       // global_addr[31:0]
  g0[3] = (unsigned)((ga >> 32) & 0x01FFFFFFu) | (2u << 30);  // [56:32], type=2
  v8i g1;
  g1[0] = (int)(1u << 16);                                    // data_size=1 (2B)
  g1[1] = (int)((tens_d0 & 0xFFFFu) << 16);                   // tensor_dim0 lo
  g1[2] = (int)(((tens_d0 >> 16) & 0xFFFFu) | ((tens_d1 & 0xFFFFu) << 16));
  g1[3] = (int)(((tens_d1 >> 16) & 0xFFFFu) | ((tile_d0 & 0xFFFFu) << 16));
  g1[4] = (int)(tile_d1 & 0xFFFFu);                           // tile_dim1, dim2=0
  g1[5] = (int)(unsigned)(stride0_elems & 0xFFFFFFFFu);       // dim0_stride lo
  g1[6] = (int)(unsigned)((stride0_elems >> 32) & 0xFFFFu);   // dim0_stride hi
  g1[7] = 0;
  v4i g2 = (v4i)0;
  v4i g3 = (v4i)0;
#if __clang_major__ >= 23
  v8i g4 = (v8i)0;
  __builtin_amdgcn_tensor_load_to_lds(g0, g1, g2, g3, g4, 0);
#else
  __builtin_amdgcn_tensor_load_to_lds(g0, g1, g2, g3, 0);
#endif
}
#endif  // BCA_TDM

// =====================================================================
// Stage 1: out = BN2(W2 @ BN1(W1 @ X)) per 128-pixel tile, WMMA bf16.
//   X: [CIN][NPIX] fp32 (channel-major, per batch)
//   outT: [NPIX][64] bf16 (pixel-major)
// Block: 256 threads = 8 waves. Wave w: out-ch tile w/2, pixel half w%2.
// =====================================================================
template<int CIN>
__global__ __launch_bounds__(256)
void bca_proj2_kernel(const float* __restrict__ X,
                      const float* __restrict__ W1, const float* __restrict__ s1,
                      const float* __restrict__ b1,
                      const float* __restrict__ W2, const float* __restrict__ s2,
                      const float* __restrict__ b2,
                      __bf16* __restrict__ outT)
{
  const int b  = blockIdx.y;
  const int n0 = blockIdx.x * TILE_N;
  X    += (size_t)b * CIN * NPIX;
  outT += (size_t)b * NPIX * CMID;

  __shared__ __bf16 Xs[TILE_N][40];     // pixel-major K-chunk (32 ch + pad)
  __shared__ __bf16 W1s[64][32];
  __shared__ __bf16 W2s[64][64];
  __shared__ __bf16 Hs[TILE_N][CMID];   // intermediate, pixel-major

  const int tid  = threadIdx.x;
  const int wave = tid >> 5;
  const int lane = tid & 31;
  const int lh   = lane >> 4;
  const int ll   = lane & 15;
  const int mtile = wave >> 1;
  const int nhalf = wave & 1;

  // W2 -> LDS bf16: 1024 float4, 4 per thread, packed v4bf stores
#pragma unroll
  for (int k = 0; k < 4; ++k) {
    const int idx = tid + 256 * k;
    const int m = idx >> 4, c = (idx & 15) << 2;
    float4 w = *(const float4*)&W2[(size_t)m * 64 + c];
    v4bf p; p[0] = f2bf(w.x); p[1] = f2bf(w.y); p[2] = f2bf(w.z); p[3] = f2bf(w.w);
    *(v4bf*)&W2s[m][c] = p;
  }

  v8f acc[4] = {};
  constexpr int NCHUNK = (CIN + 31) / 32;
  for (int ck = 0; ck < NCHUNK; ++ck) {
    const int kb = ck << 5;
    __syncthreads();
    // X chunk: 32ch x 128px = 1024 float4; branchless tail masking.
#pragma unroll
    for (int k = 0; k < 4; ++k) {
      const int idx = tid + 256 * k;
      const int c = idx >> 5;
      const int n = (idx & 31) << 2;
      const int cc = kb + c;
      const bool ok = (cc < CIN);
      float4 v = *(const float4*)&X[(size_t)(ok ? cc : 0) * NPIX + n0 + n];
      Xs[n + 0][c] = f2bf(ok ? v.x : 0.f);
      Xs[n + 1][c] = f2bf(ok ? v.y : 0.f);
      Xs[n + 2][c] = f2bf(ok ? v.z : 0.f);
      Xs[n + 3][c] = f2bf(ok ? v.w : 0.f);
    }
    // W1 chunk: 64x32 = 512 float4, 2 per thread (tail uniform per float4)
#pragma unroll
    for (int k = 0; k < 2; ++k) {
      const int idx = tid + 256 * k;
      const int m = idx >> 3;
      const int c = (idx & 7) << 2;
      const int cc = kb + c;
      const bool ok = (cc + 3 < CIN);
      float4 w = *(const float4*)&W1[(size_t)m * CIN + (ok ? cc : 0)];
      v4bf p;
      p[0] = f2bf(ok ? w.x : 0.f); p[1] = f2bf(ok ? w.y : 0.f);
      p[2] = f2bf(ok ? w.z : 0.f); p[3] = f2bf(ok ? w.w : 0.f);
      *(v4bf*)&W1s[m][c] = p;
    }
    // prefetch next K-chunk of X into cache
    if (ck + 1 < NCHUNK) {
      const int c = tid >> 3, n = (tid & 7) << 4;
      const int cc = kb + 32 + c;
      if (cc < CIN) __builtin_prefetch(&X[(size_t)cc * NPIX + n0 + n], 0, 1);
    }
    __syncthreads();

    const int row = mtile * 16 + ll;
    const v16bf a = frag16(&W1s[row][lh * 16], &W1s[row][lh * 16 + 8]);
#pragma unroll
    for (int t = 0; t < 4; ++t) {
      const int ncol = (nhalf * 4 + t) * 16 + ll;
      const v16bf bb = frag16(&Xs[ncol][lh * 16], &Xs[ncol][lh * 16 + 8]);
      acc[t] = __builtin_amdgcn_wmma_f32_16x16x32_bf16(
          false, a, false, bb, (short)0, acc[t], false, false);
    }
  }

  // BN1, stash H (bf16, pixel-major) in LDS
  const int mbase = mtile * 16 + lh * 8;
#pragma unroll
  for (int t = 0; t < 4; ++t) {
    const int ncol = (nhalf * 4 + t) * 16 + ll;
    v8bf pack;
#pragma unroll
    for (int v = 0; v < 8; ++v)
      pack[v] = f2bf(acc[t][v] * s1[mbase + v] + b1[mbase + v]);
    *(v8bf*)&Hs[ncol][mbase] = pack;
  }
  __syncthreads();

  // Stage-1b: G = W2 @ H, K = 64 (2 chunks)
  v8f acc2[4] = {};
#pragma unroll
  for (int ck = 0; ck < 2; ++ck) {
    const int row = mtile * 16 + ll;
    const v16bf a = frag16(&W2s[row][ck * 32 + lh * 16],
                           &W2s[row][ck * 32 + lh * 16 + 8]);
#pragma unroll
    for (int t = 0; t < 4; ++t) {
      const int ncol = (nhalf * 4 + t) * 16 + ll;
      const v16bf bb = frag16(&Hs[ncol][ck * 32 + lh * 16],
                              &Hs[ncol][ck * 32 + lh * 16 + 8]);
      acc2[t] = __builtin_amdgcn_wmma_f32_16x16x32_bf16(
          false, a, false, bb, (short)0, acc2[t], false, false);
    }
  }
#pragma unroll
  for (int t = 0; t < 4; ++t) {
    const int ncol = (nhalf * 4 + t) * 16 + ll;
    v8bf pack;
#pragma unroll
    for (int v = 0; v < 8; ++v)
      pack[v] = f2bf(acc2[t][v] * s2[mbase + v] + b2[mbase + v]);
    *(v8bf*)&outT[(size_t)(n0 + ncol) * CMID + mbase] = pack;
  }
}

// =====================================================================
// Stage 2: 4x4 maxpool of fself & fy (pixel-major bf16 inputs).
//   fselfP: [64][1024] channel-major (V matrix), fyT: [1024][64] key-major.
// =====================================================================
__global__ void bca_pool_kernel(const __bf16* __restrict__ fsT,
                                const __bf16* __restrict__ fyTf,
                                __bf16* __restrict__ fselfP,
                                __bf16* __restrict__ fyT)
{
  const int m = blockIdx.x, b = blockIdx.y, c = threadIdx.x;
  const int mh = m >> 5, mw = m & 31;
  const __bf16* ps = fsT  + (size_t)b * NPIX * CMID;
  const __bf16* py = fyTf + (size_t)b * NPIX * CMID;
  float ms = -3.0e38f, my = -3.0e38f;
#pragma unroll
  for (int i = 0; i < 4; ++i)
#pragma unroll
    for (int j = 0; j < 4; ++j) {
      const int n = (mh * 4 + i) * 128 + mw * 4 + j;
      ms = fmaxf(ms, bf2f(ps[(size_t)n * CMID + c]));
      my = fmaxf(my, bf2f(py[(size_t)n * CMID + c]));
    }
  fselfP[((size_t)b * CMID + c) * MKEYS + m] = f2bf(ms);
  fyT[((size_t)b * MKEYS + m) * CMID + c]    = f2bf(my);
}

// =====================================================================
// Stage 3: flash attention. Wave = 16 queries; stream 32-key chunks.
//   S = Q(16x64) K(64x32) -> online softmax -> acc += P(16x32) V(32x64)
// K/V chunks staged block-wide into LDS by the Tensor Data Mover
// (double-buffered, wave 0 issues, s_wait_tensorcnt + barrier publishes).
// =====================================================================
__global__ __launch_bounds__(256)
void bca_attn_kernel(const __bf16* __restrict__ fxT,
                     const __bf16* __restrict__ fyT,
                     const __bf16* __restrict__ fselfP,
                     __bf16* __restrict__ foutT)
{
  const int b  = blockIdx.y;
  const int q0 = blockIdx.x * 128;
  fxT    += (size_t)b * NPIX * CMID;
  foutT  += (size_t)b * NPIX * CMID;
  fyT    += (size_t)b * MKEYS * CMID;
  fselfP += (size_t)b * CMID * MKEYS;

  __shared__ __bf16 Ps[8][16][32];   // per-wave P transpose scratch
  __shared__ __bf16 Fo[8][16][64];   // per-wave output transpose scratch
#if BCA_TDM
  __shared__ __bf16 Ks[2][32][64];   // K chunk: 32 keys x 64 ch (contiguous copy)
  __shared__ __bf16 Vs[2][64][32];   // V chunk: 64 ch x 32 keys (2D strided tile)
#endif

  const int tid = threadIdx.x, wave = tid >> 5, lane = tid & 31;
  const int lh = lane >> 4, ll = lane & 15;
  const int qb = q0 + wave * 16;

  // Q A-fragments, K = 64 channels (2 chunks of 32)
  const __bf16* qp = &fxT[(size_t)(qb + ll) * CMID];
  v16bf qa[2];
#pragma unroll
  for (int ck = 0; ck < 2; ++ck)
    qa[ck] = frag16(qp + ck * 32 + lh * 16, qp + ck * 32 + lh * 16 + 8);

#if BCA_TDM
  if (wave == 0) {   // prologue: stage chunk 0 into buffer 0
    bca_tdm_load((unsigned)(unsigned long long)(const void*)&Ks[0][0][0],
                 fyT, 2048, 0, 2048, 1, 2048);
    bca_tdm_load((unsigned)(unsigned long long)(const void*)&Vs[0][0][0],
                 fselfP, 32, 64, 1024, 64, 1024);
  }
#endif

  float mrow[8], lrow[8];
#pragma unroll
  for (int v = 0; v < 8; ++v) { mrow[v] = -3.0e38f; lrow[v] = 0.f; }
  v8f acc[4] = {};
  const float LOG2E = 1.4426950408889634f;
  int cur = 0;

  for (int kb = 0; kb < MKEYS; kb += 32) {
#if BCA_TDM
    __builtin_amdgcn_s_wait_tensorcnt(0);   // buf[cur] complete (issuing wave)
    __syncthreads();                        // publish buf[cur] to all waves
    if (wave == 0 && kb + 32 < MKEYS) {     // overlap next chunk's DMA
      bca_tdm_load((unsigned)(unsigned long long)(const void*)&Ks[cur ^ 1][0][0],
                   fyT + (size_t)(kb + 32) * CMID, 2048, 0, 2048, 1, 2048);
      bca_tdm_load((unsigned)(unsigned long long)(const void*)&Vs[cur ^ 1][0][0],
                   fselfP + kb + 32, 32, 64, 1024, 64, 1024);
    }
#endif
    // scores for two 16-key tiles
    v8f S[2] = {};
#pragma unroll
    for (int kt = 0; kt < 2; ++kt) {
#if BCA_TDM
      const __bf16* kp = &Ks[cur][kt * 16 + ll][0];
#else
      const __bf16* kp = &fyT[(size_t)(kb + kt * 16 + ll) * CMID];
#endif
#pragma unroll
      for (int ck = 0; ck < 2; ++ck) {
        const v16bf bb = frag16(kp + ck * 32 + lh * 16, kp + ck * 32 + lh * 16 + 8);
        S[kt] = __builtin_amdgcn_wmma_f32_16x16x32_bf16(
            false, qa[ck], false, bb, (short)0, S[kt], false, false);
      }
    }
    // online softmax: D-tile rows live on 16-lane halves
    float corr[8];
#pragma unroll
    for (int v = 0; v < 8; ++v) {
      float t = fmaxf(S[0][v], S[1][v]);
#pragma unroll
      for (int msk = 8; msk >= 1; msk >>= 1)
        t = fmaxf(t, __shfl_xor(t, msk, 32));
      const float mn = fmaxf(mrow[v], t);
      corr[v] = exp2f((mrow[v] - mn) * LOG2E);
      mrow[v] = mn;
      const float p0 = exp2f((S[0][v] - mn) * LOG2E);
      const float p1 = exp2f((S[1][v] - mn) * LOG2E);
      S[0][v] = p0; S[1][v] = p1;
      float rs = p0 + p1;
#pragma unroll
      for (int msk = 8; msk >= 1; msk >>= 1)
        rs += __shfl_xor(rs, msk, 32);
      lrow[v] = lrow[v] * corr[v] + rs;
    }
#pragma unroll
    for (int t = 0; t < 4; ++t)
#pragma unroll
      for (int v = 0; v < 8; ++v) acc[t][v] *= corr[v];

    // D-layout -> A-layout transpose through LDS (bf16)
#pragma unroll
    for (int v = 0; v < 8; ++v) {
      Ps[wave][v + lh * 8][ll]      = f2bf(S[0][v]);
      Ps[wave][v + lh * 8][16 + ll] = f2bf(S[1][v]);
    }
    __syncthreads();
    const v16bf pa = frag16(&Ps[wave][ll][lh * 16], &Ps[wave][ll][lh * 16 + 8]);
    __syncthreads();
    // acc += P @ V
#pragma unroll
    for (int t = 0; t < 4; ++t) {
#if BCA_TDM
      const __bf16* vp = &Vs[cur][t * 16 + ll][lh * 16];
#else
      const __bf16* vp = &fselfP[(size_t)(t * 16 + ll) * MKEYS + kb + lh * 16];
#endif
      const v16bf bb = frag16(vp, vp + 8);
      acc[t] = __builtin_amdgcn_wmma_f32_16x16x32_bf16(
          false, pa, false, bb, (short)0, acc[t], false, false);
    }
    cur ^= 1;
  }

  // normalize; transpose through LDS so global stores are b128
#pragma unroll
  for (int v = 0; v < 8; ++v) {
    const float inv = 1.0f / lrow[v];
    const int r = v + lh * 8;
#pragma unroll
    for (int t = 0; t < 4; ++t)
      Fo[wave][r][t * 16 + ll] = f2bf(acc[t][v] * inv);
  }
  __syncthreads();
  {
    __bf16* dst = &foutT[(size_t)(qb + ll) * CMID + lh * 32];
    const __bf16* src = &Fo[wave][ll][lh * 32];
    *(v8bf*)(dst + 0)  = *(const v8bf*)(src + 0);
    *(v8bf*)(dst + 8)  = *(const v8bf*)(src + 8);
    *(v8bf*)(dst + 16) = *(const v8bf*)(src + 16);
    *(v8bf*)(dst + 24) = *(const v8bf*)(src + 24);
  }
}

// =====================================================================
// Stage 4: out = x + BN(Wu @ fout).  D tile: rows=pixels, cols=out-ch.
//   A = foutT (pixel-major), B = Wu^T (rows of Wu read as float4s).
// =====================================================================
__global__ __launch_bounds__(256)
void bca_up_kernel(const __bf16* __restrict__ foutT,
                   const float* __restrict__ Wu,
                   const float* __restrict__ su, const float* __restrict__ bu,
                   const float* __restrict__ x, float* __restrict__ out)
{
  const int b  = blockIdx.y;
  const int n0 = blockIdx.x * 128;
  foutT += (size_t)b * NPIX * CMID;
  x     += (size_t)b * CXIN * NPIX;
  out   += (size_t)b * CXIN * NPIX;

  const int tid = threadIdx.x, wave = tid >> 5, lane = tid & 31;
  const int lh = lane >> 4, ll = lane & 15;
  const int pb = n0 + wave * 16;   // this wave's 16-pixel tile

  const __bf16* fp = &foutT[(size_t)(pb + ll) * CMID];
  v16bf fa[2];
#pragma unroll
  for (int ck = 0; ck < 2; ++ck)
    fa[ck] = frag16(fp + ck * 32 + lh * 16, fp + ck * 32 + lh * 16 + 8);

  for (int ot = 0; ot < CXIN / 16; ++ot) {     // 45 out-channel tiles
    const int ocol = ot * 16 + ll;
    v8f acc = {};
#pragma unroll
    for (int ck = 0; ck < 2; ++ck) {
      const float* wp = &Wu[(size_t)ocol * CMID + ck * 32 + lh * 16];
      float4 w0 = *(const float4*)(wp + 0);
      float4 w1 = *(const float4*)(wp + 4);
      float4 w2 = *(const float4*)(wp + 8);
      float4 w3 = *(const float4*)(wp + 12);
      v16bf bb;
      bb[0] = f2bf(w0.x);  bb[1] = f2bf(w0.y);  bb[2] = f2bf(w0.z);  bb[3] = f2bf(w0.w);
      bb[4] = f2bf(w1.x);  bb[5] = f2bf(w1.y);  bb[6] = f2bf(w1.z);  bb[7] = f2bf(w1.w);
      bb[8] = f2bf(w2.x);  bb[9] = f2bf(w2.y);  bb[10] = f2bf(w2.z); bb[11] = f2bf(w2.w);
      bb[12] = f2bf(w3.x); bb[13] = f2bf(w3.y); bb[14] = f2bf(w3.z); bb[15] = f2bf(w3.w);
      acc = __builtin_amdgcn_wmma_f32_16x16x32_bf16(
          false, fa[ck], false, bb, (short)0, acc, false, false);
    }
    const float s = su[ocol], bi = bu[ocol];
    const size_t base = (size_t)ocol * NPIX + pb + lh * 8;
    v8f xv = *(const v8f*)&x[base];
    v8f res;
#pragma unroll
    for (int v = 0; v < 8; ++v) res[v] = xv[v] + acc[v] * s + bi;
    *(v8f*)&out[base] = res;
  }
}

// =====================================================================
extern "C" void kernel_launch(void* const* d_in, const int* in_sizes, int n_in,
                              void* d_out, int out_size, void* d_ws, size_t ws_size,
                              hipStream_t stream) {
  (void)in_sizes; (void)n_in; (void)out_size; (void)ws_size;
  const float* x   = (const float*)d_in[0];
  const float* y   = (const float*)d_in[1];
  const float* Ws1 = (const float*)d_in[2];
  const float* ss1 = (const float*)d_in[3];
  const float* bs1 = (const float*)d_in[4];
  const float* Ws2 = (const float*)d_in[5];
  const float* ss2 = (const float*)d_in[6];
  const float* bs2 = (const float*)d_in[7];
  const float* Wx1 = (const float*)d_in[8];
  const float* sx1 = (const float*)d_in[9];
  const float* bx1 = (const float*)d_in[10];
  const float* Wx2 = (const float*)d_in[11];
  const float* sx2 = (const float*)d_in[12];
  const float* bx2 = (const float*)d_in[13];
  const float* Wy1 = (const float*)d_in[14];
  const float* sy1 = (const float*)d_in[15];
  const float* by1 = (const float*)d_in[16];
  const float* Wy2 = (const float*)d_in[17];
  const float* sy2 = (const float*)d_in[18];
  const float* by2 = (const float*)d_in[19];
  const float* Wu  = (const float*)d_in[20];
  const float* su  = (const float*)d_in[21];
  const float* bu  = (const float*)d_in[22];
  float* out = (float*)d_out;

  // workspace carve-up (bf16 buffers), ~16.5 MB total
  char* ws = (char*)d_ws;
  const size_t szFull = (size_t)2 * NPIX * CMID * sizeof(__bf16);   // 4 MB
  const size_t szPool = (size_t)2 * MKEYS * CMID * sizeof(__bf16);  // 256 KB
  __bf16* fsT    = (__bf16*)(ws);                                   // fself full res
  __bf16* fxT    = (__bf16*)(ws + szFull);                          // query
  __bf16* fyTf   = (__bf16*)(ws + 2 * szFull);                      // fy full res
  __bf16* foutT  = (__bf16*)(ws + 3 * szFull);                      // attention out
  __bf16* fselfP = (__bf16*)(ws + 4 * szFull);                      // pooled V [64][1024]
  __bf16* fyT    = (__bf16*)(ws + 4 * szFull + szPool);             // pooled K [1024][64]

  const dim3 blk(256);
  const dim3 gproj(NPIX / TILE_N, 2);
  bca_proj2_kernel<CXIN><<<gproj, blk, 0, stream>>>(x, Ws1, ss1, bs1, Ws2, ss2, bs2, fsT);
  bca_proj2_kernel<CXIN><<<gproj, blk, 0, stream>>>(x, Wx1, sx1, bx1, Wx2, sx2, bx2, fxT);
  bca_proj2_kernel<CMID><<<gproj, blk, 0, stream>>>(y, Wy1, sy1, by1, Wy2, sy2, by2, fyTf);
  bca_pool_kernel<<<dim3(MKEYS, 2), dim3(64), 0, stream>>>(fsT, fyTf, fselfP, fyT);
  bca_attn_kernel<<<dim3(NPIX / 128, 2), blk, 0, stream>>>(fxT, fyT, fselfP, foutT);
  bca_up_kernel<<<dim3(NPIX / 128, 2), blk, 0, stream>>>(foutT, Wu, su, bu, x, out);
}
